// RNN_49460843381592
// MI455X (gfx1250) — matrix operand order repaired and checked by reading, not compiled, not run on hardware
//
#include <hip/hip_runtime.h>
#include <hip/hip_bf16.h>

typedef __attribute__((ext_vector_type(16))) _Float16 v16h;
typedef __attribute__((ext_vector_type(8)))  _Float16 v8h;
typedef __attribute__((ext_vector_type(8)))  float    v8f;
typedef __attribute__((ext_vector_type(4)))  unsigned int u32x4;
typedef __attribute__((ext_vector_type(8)))  unsigned int u32x8;

#define Bsz 128
#define Tsz 512
#define Dsz 64
#define Hsz 512
#define Csz 5

// ---------------------------------------------------------------------------
// WMMA fragment helpers (wave32, 16x16x32 f16 -> f32)
//  A (16x32, M x K): lane m = lane&15, hi = lane>>4.
//    a[i] (i<8)  -> K = hi*16 + i           (contiguous 16B)
//    a[i] (i>=8) -> K = hi*16 + i           (contiguous 16B)
//  B (32x16, K x N) with B[k,n] = W[n,k] (W row-major):
//    b[i] -> K = kk + i, N = lane&15, kk = kc*32 + (lane>>4)*16  (contiguous 32B)
//  C/D f32: c[r] -> M = r + 8*(lane>>4), N = lane&15
// ---------------------------------------------------------------------------
__device__ __forceinline__ v16h load_A16(const _Float16* rowk, int hi) {
  union { v16h v; v8h h[2]; } u;
  u.h[0] = *(const v8h*)(rowk + hi * 16);
  u.h[1] = *(const v8h*)(rowk + hi * 16 + 8);
  return u.v;
}

// ---------------------------------------------------------------------------
// fp32 -> f16 conversion (grid-stride)
// ---------------------------------------------------------------------------
__global__ __launch_bounds__(256) void cvt_f32_to_f16(const float* __restrict__ src,
                                                      _Float16* __restrict__ dst, int n) {
  for (int i = blockIdx.x * 256 + threadIdx.x; i < n; i += gridDim.x * 256)
    dst[i] = (_Float16)src[i];
}

// ---------------------------------------------------------------------------
// xp = A(f16, M x K row-major) @ Bw(f16, N=512 rows x K)^T + bias   -> fp32 (M x 512)
// Each wave: one 16-row M tile x 64 N columns (4 WMMA tiles, A reused x4).
// ---------------------------------------------------------------------------
template <int K>
__global__ __launch_bounds__(256) void gemm_xp(const _Float16* __restrict__ A,
                                               const _Float16* __restrict__ Bw,
                                               const float* __restrict__ bias,
                                               float* __restrict__ out) {
  const int lane = threadIdx.x & 31;
  const int wave = threadIdx.x >> 5;
  const int gw   = blockIdx.x * 8 + wave;
  const int ng   = gw & 7;        // N-group (64 cols)
  const int mt   = gw >> 3;       // M tile
  const int M0   = mt * 16;
  const int n0w  = ng * 64;
  const int m  = lane & 15;
  const int hi = lane >> 4;
  const int n  = lane & 15;

  const _Float16* Arow = A + (size_t)(M0 + m) * K;

  v8f acc[4];
#pragma unroll
  for (int nt = 0; nt < 4; ++nt) {
    float bz = bias[n0w + nt * 16 + n];
#pragma unroll
    for (int r = 0; r < 8; ++r) acc[nt][r] = bz;
  }

#pragma unroll
  for (int kc = 0; kc < K / 32; ++kc) {
    v16h a = load_A16(Arow + kc * 32, hi);
#pragma unroll
    for (int nt = 0; nt < 4; ++nt) {
      v16h bf = *(const v16h*)(Bw + (size_t)(n0w + nt * 16 + n) * K + kc * 32 + hi * 16);
      acc[nt] = __builtin_amdgcn_wmma_f32_16x16x32_f16(false, a, false, bf,
                                                       (short)0, acc[nt], false, false);
    }
  }

#pragma unroll
  for (int nt = 0; nt < 4; ++nt)
#pragma unroll
    for (int r = 0; r < 8; ++r) {
      int mm = r + 8 * hi;
      out[(size_t)(M0 + mm) * Hsz + n0w + nt * 16 + n] = acc[nt][r];
    }
}

// ---------------------------------------------------------------------------
// TDM (iterated): stage `W_hh` rows into LDS with a padded row stride.
// Each iteration moves one row of NKL*32 halves; lds_addr advances by
// WLDS_LD halves per iteration (the pad), global_addr by Hsz halves.
// 256 iterations per descriptor (iterate_count max) -> two issues per fill.
// ---------------------------------------------------------------------------
#define NKL     9     // # of K-chunks (of 32) resident in LDS
#define WLDS_LD 296   // W LDS row stride in halves (NKL*32 + 8 pad)

__device__ __forceinline__ void tdm_fill_w256(const _Float16* gsrc, const _Float16* ldsDst) {
  unsigned long long ga = (unsigned long long)(const void*)gsrc;
  unsigned lds = (unsigned)(unsigned long long)(const void*)ldsDst;
  u32x4 g0;
  g0[0] = 1u;                                                  // count=1 (valid D#)
  g0[1] = lds;                                                 // lds_addr
  g0[2] = (unsigned)ga;                                        // global_addr[31:0]
  g0[3] = (unsigned)((ga >> 32) & 0x1FFFFFFull) | (2u << 30);  // addr[56:32], type=2
  u32x8 g1;
  g1[0] = (1u << 16) | (1u << 19);   // data_size=2B, iterate_enable=1
  g1[1] = ((unsigned)Hsz << 16);     // tensor_dim0 = 512 halves
  g1[2] = (512u << 16);              // tensor_dim1
  g1[3] = ((unsigned)(NKL * 32) << 16); // tile_dim0 = 288 halves
  g1[4] = 1u;                        // tile_dim1 = 1 (one row per iteration)
  g1[5] = (unsigned)Hsz;             // tensor_dim0_stride = 512
  g1[6] = 0u;
  g1[7] = 0u;
  u32x4 g2;
  g2[0] = 0u;                        // tensor_dim2 (unused)
  g2[1] = (unsigned)WLDS_LD;         // lds_addr_increment (data_size units)
  g2[2] = (unsigned)Hsz;             // global_addr_increment (data_size units)
  g2[3] = (255u << 16);              // iterate_count = 255 -> 256 iterations
  u32x4 g3;
  g3[0] = 0u; g3[1] = 0u; g3[2] = 0u; g3[3] = 0u;
  asm volatile("tensor_load_to_lds %0, %1, %2, %3"
               :: "s"(g0), "s"(g1), "s"(g2), "s"(g3) : "memory");
}

// ---------------------------------------------------------------------------
// Recurrence: h_t = relu(xp_t + h_{t-1} @ Whh^T) for one layer.
// 8 workgroups, each owns 16 batch rows; h-state (16x512 f16) lives in LDS.
//  - W_hh K-chunks 0..8 staged once in LDS by the TDM (303 KB), reused 512x
//  - K-chunks 9..15 streamed from L2 with a 2-deep register double-buffer;
//    loads refill a slot AFTER its consuming WMMAs (no register copies)
//  - GEMM starts from C=0 (inline-zero C); xp is added in the epilogue from
//    32 plain global b32 loads issued at the top of the step (latency hidden
//    under the whole K-loop, values land directly in scalar regs, no moves)
//  - all addressing hoisted out of the t-loop (pointers advance by +Hsz)
// ---------------------------------------------------------------------------
#define HA_LD 520   // h-state row stride (halves), pad to spread LDS banks

__global__ __launch_bounds__(256) void rnn_recurrence(const float* __restrict__ xp,
                                                      const _Float16* __restrict__ Whh,
                                                      _Float16* __restrict__ hout) {
  __shared__ _Float16 hA[16 * HA_LD];        //  16,640 B
  __shared__ _Float16 Wlds[512 * WLDS_LD];   // 303,104 B   (total 319,744 B)

  const int tid  = threadIdx.x;
  const int lane = tid & 31;
  const int wave = tid >> 5;
  const int b0   = blockIdx.x * 16;
  const int m  = lane & 15;
  const int hi = lane >> 4;
  const int n  = lane & 15;
  const int n0w = wave * 64;

  // h_0 = 0
  for (int i = tid; i < 16 * HA_LD; i += 256) hA[i] = (_Float16)0.f;

  if (wave == 0) {  // TDM-stage W_hh[0..511][0..287] into LDS (padded rows)
    tdm_fill_w256(Whh, Wlds);
    tdm_fill_w256(Whh + 256 * Hsz, Wlds + 256 * WLDS_LD);
    __builtin_amdgcn_s_wait_tensorcnt(0);
  }
  __syncthreads();

  // ---- loop-invariant addressing (per lane) ----
  const _Float16* wg[4];   // global W_hh fragment bases (offset kc*32 halves)
  const _Float16* wl[4];   // LDS W fragment bases
#pragma unroll
  for (int nt = 0; nt < 4; ++nt) {
    wg[nt] = Whh + (size_t)(n0w + nt * 16 + n) * Hsz + hi * 16;
    wl[nt] = &Wlds[(n0w + nt * 16 + n) * WLDS_LD + hi * 16];
  }
  int        hAoff[8];     // hA store index per C-row r (+ nt*16)
  _Float16*  hrow[8];      // hout row pointer per C-row r (advances by Hsz)
  const float* xrow[8];    // xp  row pointer per C-row r (advances by Hsz)
#pragma unroll
  for (int r = 0; r < 8; ++r) {
    const int mm = r + 8 * hi;
    hAoff[r] = mm * HA_LD + n0w + n;
    hrow[r]  = hout + ((size_t)(b0 + mm) * Tsz) * Hsz + n0w + n;
    xrow[r]  = xp   + ((size_t)(b0 + mm) * Tsz) * Hsz + n0w + n;
  }
  const _Float16* ArowBase = &hA[m * HA_LD];
  const v8f vzero = {};

#pragma clang loop unroll(disable)
  for (int t = 0; t < Tsz; ++t) {
    // xp for this step: 32 direct global loads, consumed only after the
    // K-loop + barrier (latency hidden under ~64 WMMAs)
    float xg[4][8];
#pragma unroll
    for (int nt = 0; nt < 4; ++nt)
#pragma unroll
      for (int r = 0; r < 8; ++r)
        xg[nt][r] = xrow[r][nt * 16];

    // prefetch global B-chunks kc = NKL, NKL+1 (2-deep rolling buffer)
    v16h bg[2][4];
#pragma unroll
    for (int s = 0; s < 2; ++s)
#pragma unroll
      for (int nt = 0; nt < 4; ++nt)
        bg[s][nt] = *(const v16h*)(wg[nt] + (NKL + s) * 32);

    v8f acc[4];
    // K-chunk 0 from LDS with inline-zero C
    {
      v16h a = load_A16(ArowBase, hi);
#pragma unroll
      for (int nt = 0; nt < 4; ++nt)
        acc[nt] = __builtin_amdgcn_wmma_f32_16x16x32_f16(false, a, false,
                                                         *(const v16h*)(wl[nt]),
                                                         (short)0, vzero, false, false);
    }
    // K-chunks 1..8 from LDS (global loads above land meanwhile)
#pragma unroll
    for (int kc = 1; kc < NKL; ++kc) {
      v16h a = load_A16(ArowBase + kc * 32, hi);
#pragma unroll
      for (int nt = 0; nt < 4; ++nt) {
        v16h bf = *(const v16h*)(wl[nt] + kc * 32);
        acc[nt] = __builtin_amdgcn_wmma_f32_16x16x32_f16(false, a, false, bf,
                                                         (short)0, acc[nt], false, false);
      }
    }
    // K-chunks 9..15 from L2; consume bg[slot] directly, then refill the slot
    // with chunk kc+2 (load lands two chunks ahead of its use; no reg copies)
#pragma unroll
    for (int kc = NKL; kc < 16; ++kc) {
      const int slot = (kc - NKL) & 1;
      v16h a = load_A16(ArowBase + kc * 32, hi);
#pragma unroll
      for (int nt = 0; nt < 4; ++nt)
        acc[nt] = __builtin_amdgcn_wmma_f32_16x16x32_f16(false, a, false, bg[slot][nt],
                                                         (short)0, acc[nt], false, false);
      if (kc + 2 < 16) {
#pragma unroll
        for (int nt = 0; nt < 4; ++nt)
          bg[slot][nt] = *(const v16h*)(wg[nt] + (kc + 2) * 32);
      }
    }

    __syncthreads();  // everyone done reading hA

    // epilogue: h = relu(acc + xp), write h-state (LDS) + hout (global)
#pragma unroll
    for (int nt = 0; nt < 4; ++nt)
#pragma unroll
      for (int r = 0; r < 8; ++r) {
        float v = acc[nt][r] + xg[nt][r];
        v = v > 0.f ? v : 0.f;  // relu
        _Float16 hv = (_Float16)v;
        hA[hAoff[r] + nt * 16] = hv;
        hrow[r][nt * 16] = hv;
      }
#pragma unroll
    for (int r = 0; r < 8; ++r) { hrow[r] += Hsz; xrow[r] += Hsz; }

    __syncthreads();  // hA(t) visible to all waves for step t+1
  }
}

// ---------------------------------------------------------------------------
// out[b, c] = b_out[c] + sum_k h16[b, k] * W_out[c, k]  (K = T*H = 262144)
// ---------------------------------------------------------------------------
__global__ __launch_bounds__(256) void final_linear(const _Float16* __restrict__ h16,
                                                    const float* __restrict__ Wout,
                                                    const float* __restrict__ bout,
                                                    float* __restrict__ out) {
  const int b = blockIdx.x;
  const int tid = threadIdx.x;
  const int K = Tsz * Hsz;
  const _Float16* hb = h16 + (size_t)b * K;

  float acc[Csz] = {0.f, 0.f, 0.f, 0.f, 0.f};
  for (int k = tid; k < K; k += 256) {
    float hv = (float)hb[k];
#pragma unroll
    for (int c = 0; c < Csz; ++c) acc[c] += hv * Wout[(size_t)c * K + k];
  }

  __shared__ float red[Csz][256];
#pragma unroll
  for (int c = 0; c < Csz; ++c) red[c][tid] = acc[c];
  __syncthreads();
  for (int s = 128; s > 0; s >>= 1) {
    if (tid < s)
#pragma unroll
      for (int c = 0; c < Csz; ++c) red[c][tid] += red[c][tid + s];
    __syncthreads();
  }
  if (tid < Csz) out[b * Csz + tid] = red[tid][0] + bout[tid];
}

// ---------------------------------------------------------------------------
// host-side launcher
// ---------------------------------------------------------------------------
extern "C" void kernel_launch(void* const* d_in, const int* in_sizes, int n_in,
                              void* d_out, int out_size, void* d_ws, size_t ws_size,
                              hipStream_t stream) {
  (void)in_sizes; (void)n_in; (void)out_size; (void)ws_size;

  const float* x      = (const float*)d_in[0];  // (B,T,D)
  const float* W_ih0  = (const float*)d_in[1];  // (H,D)
  const float* W_hh0  = (const float*)d_in[2];  // (H,H)
  const float* b0     = (const float*)d_in[3];  // (H)
  const float* W_ihr  = (const float*)d_in[4];  // (2,H,H)
  const float* W_hhr  = (const float*)d_in[5];  // (2,H,H)
  const float* b_rest = (const float*)d_in[6];  // (2,H)
  const float* W_out  = (const float*)d_in[7];  // (C, T*H)
  const float* b_out  = (const float*)d_in[8];  // (C)
  float* out = (float*)d_out;

  // workspace layout (bytes)
  char* ws = (char*)d_ws;
  const size_t XP_OFF   = 0;                                   // B*T*H fp32 = 128 MiB
  const size_t H16_OFF  = XP_OFF  + (size_t)Bsz*Tsz*Hsz*4;     // B*T*H f16  =  64 MiB
  const size_t X16_OFF  = H16_OFF + (size_t)Bsz*Tsz*Hsz*2;     // B*T*D f16  =   8 MiB
  const size_t WIH0_OFF = X16_OFF + (size_t)Bsz*Tsz*Dsz*2;     // 512*64 f16
  const size_t WHH_OFF  = WIH0_OFF + (size_t)Hsz*Dsz*2;        // 3*512*512 f16
  const size_t WIHR_OFF = WHH_OFF + 3ull*Hsz*Hsz*2;            // 2*512*512 f16

  float*    xp    = (float*)   (ws + XP_OFF);
  _Float16* h16   = (_Float16*)(ws + H16_OFF);
  _Float16* x16   = (_Float16*)(ws + X16_OFF);
  _Float16* wih0  = (_Float16*)(ws + WIH0_OFF);
  _Float16* whh   = (_Float16*)(ws + WHH_OFF);   // [3][H*H]
  _Float16* wihr  = (_Float16*)(ws + WIHR_OFF);  // [2][H*H]

  // --- fp32 -> f16 conversions ---
  cvt_f32_to_f16<<<1024, 256, 0, stream>>>(x, x16, Bsz*Tsz*Dsz);
  cvt_f32_to_f16<<<64,   256, 0, stream>>>(W_ih0, wih0, Hsz*Dsz);
  cvt_f32_to_f16<<<256,  256, 0, stream>>>(W_hh0, whh, Hsz*Hsz);
  cvt_f32_to_f16<<<512,  256, 0, stream>>>(W_hhr, whh + (size_t)Hsz*Hsz, 2*Hsz*Hsz);
  cvt_f32_to_f16<<<512,  256, 0, stream>>>(W_ihr, wihr, 2*Hsz*Hsz);

  const int gemm_blocks = (Bsz*Tsz/16) * 8 / 8;  // 4096

  // --- layer 0 ---
  gemm_xp<Dsz><<<gemm_blocks, 256, 0, stream>>>(x16, wih0, b0, xp);
  rnn_recurrence<<<Bsz/16, 256, 0, stream>>>(xp, whh, h16);

  // --- layers 1..2 ---
  for (int l = 0; l < 2; ++l) {
    gemm_xp<Hsz><<<gemm_blocks, 256, 0, stream>>>(h16, wihr + (size_t)l*Hsz*Hsz,
                                                  b_rest + l*Hsz, xp);
    rnn_recurrence<<<Bsz/16, 256, 0, stream>>>(xp, whh + (size_t)(l+1)*Hsz*Hsz, h16);
  }

  // --- final linear ---
  final_linear<<<Bsz, 256, 0, stream>>>(h16, W_out, b_out, out);
}
